// CausalSelfAttention_40123584479483
// MI455X (gfx1250) — matrix operand order repaired and checked by reading, not compiled, not run on hardware
//
#include <hip/hip_runtime.h>

typedef __bf16 bf16;
typedef __attribute__((ext_vector_type(16))) __bf16 v16bf;
typedef __attribute__((ext_vector_type(8)))  float  v8f;

typedef unsigned int u32x4 __attribute__((ext_vector_type(4)));
typedef int          i32x8 __attribute__((ext_vector_type(8)));
typedef int          i32x4 __attribute__((ext_vector_type(4)));

#define B_DIM 4
#define T_DIM 2048
#define C_DIM 1024
#define H_DIM 16
#define D_DIM 64

__device__ __forceinline__ bf16 to_bf16(float f) { return (bf16)f; }

// Build a 16-element bf16 fragment from two 16-byte chunks.
__device__ __forceinline__ v16bf load_frag16(const bf16* p0, const bf16* p1) {
    union { v16bf v; uint4 u[2]; } r;
    r.u[0] = *reinterpret_cast<const uint4*>(p0);
    r.u[1] = *reinterpret_cast<const uint4*>(p1);
    return r.v;
}

__device__ __forceinline__ v8f wmma_bf16(v16bf a, v16bf b, v8f c) {
    return __builtin_amdgcn_wmma_f32_16x16x32_bf16(false, a, false, b, (short)0, c, false, false);
}

#if __has_builtin(__builtin_amdgcn_tensor_load_to_lds)
#define CSA_USE_TDM 1
#else
#define CSA_USE_TDM 0
#endif

#if CSA_USE_TDM
// TDM 2-D tile load: bf16 elements, tile (tile_d1 rows x tile_d0 elems),
// row stride = stride_elems. Tile start = gaddr; tensor dims set equal to the
// tile (with padded row stride) so the transfer is in-bounds by construction.
// D# bitfields per CDNA5 ISA ch.8 (group0 128b, group1 256b; groups 2/3 zero).
// This toolchain exposes the 6-arg builtin: (g0, g1, g2, g3, g4, cpol).
__device__ __forceinline__ void tdm_load_2d_bf16(unsigned int lds_off, const void* gaddr,
                                                 unsigned int tile_d0, unsigned int tile_d1,
                                                 unsigned int stride_elems) {
    unsigned long long ga = (unsigned long long)(uintptr_t)gaddr;
    u32x4 g0;
    g0[0] = 1u;                                   // count=1 (valid), user mode, no gather
    g0[1] = lds_off;                              // lds_addr (bytes)
    g0[2] = (unsigned int)ga;                     // global_addr[31:0]
    g0[3] = (unsigned int)((ga >> 32) & 0x01FFFFFFu) | 0x80000000u; // addr[56:32] | type=2
    i32x8 g1;
    g1[0] = 0x00010000;                           // wg_mask=0, data_size=1 (2 bytes)
    g1[1] = (int)((tile_d0 & 0xFFFFu) << 16);     // tensor_dim0[15:0] @ bits 63:48
    g1[2] = (int)((tile_d0 >> 16) | ((tile_d1 & 0xFFFFu) << 16)); // dim0 hi | dim1 lo
    g1[3] = (int)((tile_d1 >> 16) | (tile_d0 << 16));             // dim1 hi | tile_dim0
    g1[4] = (int)(tile_d1 & 0xFFFFu);             // tile_dim1 | tile_dim2=0
    g1[5] = (int)stride_elems;                    // tensor_dim0_stride[31:0]
    g1[6] = 0;                                    // stride hi16 | dim1_stride lo16
    g1[7] = 0;
    i32x4 z4 = {0, 0, 0, 0};
    i32x8 z8 = {0, 0, 0, 0, 0, 0, 0, 0};
    __builtin_amdgcn_tensor_load_to_lds(g0, g1, z4, z4, z8, 0);
}
#endif

// ---------------- conversion kernels ----------------

__global__ __launch_bounds__(256) void csa_cvt_bf16(const float* __restrict__ in,
                                                    bf16* __restrict__ out, int n) {
    int i = blockIdx.x * 256 + threadIdx.x;
    if (i < n) out[i] = to_bf16(in[i]);
}

// in: [K][N] f32  ->  out: [N][K] bf16 (transpose + convert)
__global__ __launch_bounds__(256) void csa_cvt_transpose(const float* __restrict__ in,
                                                         bf16* __restrict__ out,
                                                         int K, int N) {
    int idx = blockIdx.x * 256 + threadIdx.x;
    if (idx < K * N) {
        int k = idx / N;
        int n = idx - k * N;
        out[(size_t)n * K + k] = to_bf16(in[idx]);
    }
}

// ---------------- shared GEMM mainloop ----------------
// A: [M][K] bf16 row-major. Bt: [N][K] bf16 (B transposed). Each wave computes
// a 16(M) x 64(N) strip as four 16x16 f32 accumulators. The 64x32 weight tile
// is staged into LDS by the Tensor Data Mover, double-buffered: wave 0 issues
// the DMA for tile k+1 while all waves compute on tile k.
__device__ __forceinline__ void gemm_tile_loop(const bf16* __restrict__ A,
                                               const bf16* __restrict__ Bt,
                                               int K, int m0, int n0,
                                               bf16* lds_b /*2 * 64*32*/, v8f acc[4]) {
    const int tid  = threadIdx.x;
    const int wave = tid >> 5;
    const int lane = tid & 31;
    const int half = lane >> 4;
    const int lnr  = lane & 15;
    const int NK   = K >> 5;                       // K / 32 tiles

#if CSA_USE_TDM
    const unsigned int lds_base = (unsigned int)(uintptr_t)(void*)lds_b;
    if (wave == 0) {   // prologue: stage tile 0 into buffer 0
        tdm_load_2d_bf16(lds_base, Bt + (size_t)n0 * K, 32u, 64u, (unsigned int)K);
    }
    for (int kk = 0; kk < NK; ++kk) {
        if (wave == 0) {
            if (kk + 1 < NK) {   // issue next tile into the other buffer
                tdm_load_2d_bf16(lds_base + (unsigned int)(((kk + 1) & 1) * 64 * 32 * 2),
                                 Bt + (size_t)n0 * K + (kk + 1) * 32, 32u, 64u,
                                 (unsigned int)K);
                __builtin_amdgcn_s_wait_tensorcnt(1);   // tile kk landed
            } else {
                __builtin_amdgcn_s_wait_tensorcnt(0);
            }
        }
        __syncthreads();                                // publish tile kk

        const bf16* lb = lds_b + (kk & 1) * (64 * 32);
        const bf16* ap = A + (size_t)(m0 + lnr) * K + kk * 32 + half * 16;
        v16bf af = load_frag16(ap, ap + 8);             // A frag (contiguous K)
#pragma unroll
        for (int ns = 0; ns < 4; ++ns) {                // B frag: contiguous K @ half*16
            const bf16* bp = &lb[(ns * 16 + lnr) * 32 + half * 16];
            v16bf bfr = load_frag16(bp, bp + 8);
            acc[ns] = wmma_bf16(af, bfr, acc[ns]);
        }
        __syncthreads();                                // release buffer (kk&1)
    }
#else
    for (int kk = 0; kk < NK; ++kk) {
        const int k0 = kk * 32;
        {   // cooperative stage of 64(N) x 32(K) weight tile, 16B per thread
            const int r = tid >> 2;
            const int c = (tid & 3) * 8;
            *reinterpret_cast<uint4*>(&lds_b[r * 32 + c]) =
                *reinterpret_cast<const uint4*>(&Bt[(size_t)(n0 + r) * K + k0 + c]);
        }
        __syncthreads();
        const bf16* ap = A + (size_t)(m0 + lnr) * K + k0 + half * 16;
        v16bf af = load_frag16(ap, ap + 8);
#pragma unroll
        for (int ns = 0; ns < 4; ++ns) {
            const bf16* bp = &lds_b[(ns * 16 + lnr) * 32 + half * 16];
            v16bf bfr = load_frag16(bp, bp + 8);
            acc[ns] = wmma_bf16(af, bfr, acc[ns]);
        }
        __syncthreads();
    }
#endif
}

// ---------------- QKV GEMM: x[8192,1024] @ w_attn -> Q/K/Vt (head-major bf16) ----
__global__ __launch_bounds__(256) void csa_gemm_qkv(const bf16* __restrict__ A,
                                                    const bf16* __restrict__ Bt,
                                                    const float* __restrict__ bias,
                                                    bf16* __restrict__ Qo,
                                                    bf16* __restrict__ Ko,
                                                    bf16* __restrict__ Vto) {
    __shared__ bf16 lds_b[2 * 64 * 32];
    const int tid  = threadIdx.x;
    const int wave = tid >> 5;
    const int lane = tid & 31;
    const int half = lane >> 4;
    const int lnr  = lane & 15;
    const int m0 = blockIdx.y * 128 + wave * 16;
    const int n0 = blockIdx.x * 64;

    v8f acc[4];
#pragma unroll
    for (int i = 0; i < 4; ++i)
#pragma unroll
        for (int r = 0; r < 8; ++r) acc[i][r] = 0.f;

    gemm_tile_loop(A, Bt, C_DIM, m0, n0, lds_b, acc);

#pragma unroll
    for (int ns = 0; ns < 4; ++ns) {
#pragma unroll
        for (int r = 0; r < 8; ++r) {
            const int m = m0 + r + half * 8;           // row = b*T + t
            const int n = n0 + ns * 16 + lnr;          // col in [0, 3072)
            const float v = acc[ns][r] + bias[n];
            const bf16 bv = to_bf16(v);
            const int b = m >> 11;                     // T = 2048
            const int t = m & (T_DIM - 1);
            if (n < C_DIM) {
                const int h = n >> 6, d = n & 63;
                Qo[(((size_t)(b * H_DIM + h)) * T_DIM + t) * D_DIM + d] = bv;
            } else if (n < 2 * C_DIM) {
                const int nn = n - C_DIM;
                const int h = nn >> 6, d = nn & 63;
                Ko[(((size_t)(b * H_DIM + h)) * T_DIM + t) * D_DIM + d] = bv;
            } else {
                const int nn = n - 2 * C_DIM;
                const int h = nn >> 6, d = nn & 63;    // V transposed: [b,h,d,t]
                Vto[(((size_t)(b * H_DIM + h)) * D_DIM + d) * T_DIM + t] = bv;
            }
        }
    }
}

// ---------------- flash-style causal attention -------------------------------
// Grid: (B*H, T/128). 8 waves x 16 queries each. Streams 32-key tiles with
// online softmax; P round-trips through per-wave LDS to reach A-fragment layout.
__global__ __launch_bounds__(256) void csa_attn(const bf16* __restrict__ Q,
                                                const bf16* __restrict__ Km,
                                                const bf16* __restrict__ Vt,
                                                bf16* __restrict__ att) {
    __shared__ bf16 lds_p[8][16 * 32];
    const int tid  = threadIdx.x;
    const int wave = tid >> 5;
    const int lane = tid & 31;
    const int half = lane >> 4;
    const int lnr  = lane & 15;
    const int bh = blockIdx.x;
    const int b  = bh >> 4;
    const int h  = bh & (H_DIM - 1);
    const int q0 = blockIdx.y * 128 + wave * 16;

    const bf16* Qbh = Q  + (size_t)bh * T_DIM * D_DIM;   // [T][D]
    const bf16* Kbh = Km + (size_t)bh * T_DIM * D_DIM;   // [T][D]
    const bf16* Vbh = Vt + (size_t)bh * D_DIM * T_DIM;   // [D][T]

    const float NEG_INF = -__builtin_inff();

    // Q A-fragments for the two 32-wide d chunks (reused across all key tiles)
    v16bf qa[2];
#pragma unroll
    for (int dc = 0; dc < 2; ++dc) {
        const bf16* p = Qbh + (size_t)(q0 + lnr) * D_DIM + dc * 32 + half * 16;
        qa[dc] = load_frag16(p, p + 8);
    }

    float m_row[8], l_row[8];
    v8f accv[4];
#pragma unroll
    for (int r = 0; r < 8; ++r) { m_row[r] = NEG_INF; l_row[r] = 0.f; }
#pragma unroll
    for (int dt = 0; dt < 4; ++dt)
#pragma unroll
        for (int r = 0; r < 8; ++r) accv[dt][r] = 0.f;

    const float scale = 0.125f;                // 1/sqrt(64)
    bf16* pbuf = &lds_p[wave][0];
    const int kend = q0 + 16;                  // last key needed (exclusive)

    for (int kt = 0; kt < kend; kt += 32) {
        // ---- S = Q @ K^T for two 16-key halves ----
        v8f s[2];
#pragma unroll
        for (int j = 0; j < 2; ++j) {
#pragma unroll
            for (int r = 0; r < 8; ++r) s[j][r] = 0.f;
#pragma unroll
            for (int dc = 0; dc < 2; ++dc) {
                const bf16* kp = Kbh + (size_t)(kt + j * 16 + lnr) * D_DIM + dc * 32 + half * 16;
                v16bf kf = load_frag16(kp, kp + 8);
                s[j] = wmma_bf16(qa[dc], kf, s[j]);
            }
        }

        // ---- scale + causal mask + row max (16-lane reduction) ----
        float tmax[8];
#pragma unroll
        for (int r = 0; r < 8; ++r) {
            const int q = q0 + r + half * 8;
            float best = NEG_INF;
#pragma unroll
            for (int j = 0; j < 2; ++j) {
                const int key = kt + j * 16 + lnr;
                float v = s[j][r] * scale;
                v = (key <= q) ? v : NEG_INF;
                s[j][r] = v;
                best = fmaxf(best, v);
            }
#pragma unroll
            for (int mk = 1; mk < 16; mk <<= 1)
                best = fmaxf(best, __shfl_xor(best, mk, 32));
            tmax[r] = best;
        }

        // ---- online softmax update; write P (bf16) to per-wave LDS ----
        float alpha[8], tsum[8];
#pragma unroll
        for (int r = 0; r < 8; ++r) {
            const float mn = fmaxf(m_row[r], tmax[r]);
            alpha[r] = (mn == NEG_INF) ? 1.f : __expf(m_row[r] - mn);
            m_row[r] = mn;
            const int m = r + half * 8;
            float ssum = 0.f;
#pragma unroll
            for (int j = 0; j < 2; ++j) {
                const float p = (m_row[r] == NEG_INF) ? 0.f : __expf(s[j][r] - m_row[r]);
                ssum += p;
                pbuf[m * 32 + j * 16 + lnr] = to_bf16(p);
            }
#pragma unroll
            for (int mk = 1; mk < 16; mk <<= 1)
                ssum += __shfl_xor(ssum, mk, 32);
            tsum[r] = ssum;
        }
#pragma unroll
        for (int r = 0; r < 8; ++r) l_row[r] = l_row[r] * alpha[r] + tsum[r];
#pragma unroll
        for (int dt = 0; dt < 4; ++dt)
#pragma unroll
            for (int r = 0; r < 8; ++r) accv[dt][r] *= alpha[r];

        // DS ops from one wave complete in order; wait ensures stores landed
        asm volatile("s_wait_dscnt 0" ::: "memory");

        // ---- O += P @ V ----
        const bf16* pr = pbuf + lnr * 32 + half * 16;
        v16bf pf = load_frag16(pr, pr + 8);
#pragma unroll
        for (int dt = 0; dt < 4; ++dt) {
            const bf16* vp = Vbh + (size_t)(dt * 16 + lnr) * T_DIM + kt + half * 16;
            v16bf vf = load_frag16(vp, vp + 8);
            accv[dt] = wmma_bf16(pf, vf, accv[dt]);
        }
    }

    // ---- normalize + store att[b,t, h*64+d] as bf16 ----
#pragma unroll
    for (int r = 0; r < 8; ++r) {
        const float inv = 1.f / l_row[r];
        const int q = q0 + r + half * 8;
#pragma unroll
        for (int dt = 0; dt < 4; ++dt) {
            const int col = h * D_DIM + dt * 16 + lnr;
            att[((size_t)(b * T_DIM + q)) * C_DIM + col] = to_bf16(accv[dt][r] * inv);
        }
    }
}

// ---------------- output projection GEMM (f32 epilogue into d_out) ----------
__global__ __launch_bounds__(256) void csa_gemm_proj(const bf16* __restrict__ A,
                                                     const bf16* __restrict__ Bt,
                                                     const float* __restrict__ bias,
                                                     float* __restrict__ out) {
    __shared__ bf16 lds_b[2 * 64 * 32];
    const int tid  = threadIdx.x;
    const int wave = tid >> 5;
    const int lane = tid & 31;
    const int half = lane >> 4;
    const int lnr  = lane & 15;
    const int m0 = blockIdx.y * 128 + wave * 16;
    const int n0 = blockIdx.x * 64;

    v8f acc[4];
#pragma unroll
    for (int i = 0; i < 4; ++i)
#pragma unroll
        for (int r = 0; r < 8; ++r) acc[i][r] = 0.f;

    gemm_tile_loop(A, Bt, C_DIM, m0, n0, lds_b, acc);

#pragma unroll
    for (int ns = 0; ns < 4; ++ns) {
#pragma unroll
        for (int r = 0; r < 8; ++r) {
            const int m = m0 + r + half * 8;
            const int n = n0 + ns * 16 + lnr;
            out[(size_t)m * C_DIM + n] = acc[ns][r] + bias[n];
        }
    }
}

// ---------------- host-side launch ----------------

extern "C" void kernel_launch(void* const* d_in, const int* in_sizes, int n_in,
                              void* d_out, int out_size, void* d_ws, size_t ws_size,
                              hipStream_t stream) {
    (void)in_sizes; (void)n_in; (void)out_size; (void)ws_size;

    const float* x      = (const float*)d_in[0];   // [8192, 1024]
    const float* w_attn = (const float*)d_in[1];   // [1024, 3072]
    const float* b_attn = (const float*)d_in[2];   // [3072]
    const float* w_proj = (const float*)d_in[3];   // [1024, 1024]
    const float* b_proj = (const float*)d_in[4];   // [1024]
    float* out = (float*)d_out;                    // [8192, 1024]

    char* ws = (char*)d_ws;
    bf16* xb     = (bf16*)ws; ws += (size_t)8192 * 1024 * 2;   // 16 MB
    bf16* wqkvT  = (bf16*)ws; ws += (size_t)3072 * 1024 * 2;   //  6 MB
    bf16* wprojT = (bf16*)ws; ws += (size_t)1024 * 1024 * 2;   //  2 MB
    bf16* Qb     = (bf16*)ws; ws += (size_t)8192 * 1024 * 2;   // 16 MB  [b,h,t,d]
    bf16* Kb     = (bf16*)ws; ws += (size_t)8192 * 1024 * 2;   // 16 MB  [b,h,t,d]
    bf16* Vt     = (bf16*)ws; ws += (size_t)8192 * 1024 * 2;   // 16 MB  [b,h,d,t]
    bf16* attb   = (bf16*)ws; ws += (size_t)8192 * 1024 * 2;   // 16 MB  [b*t, c]

    // precision/layout prep
    csa_cvt_bf16<<<(8192 * 1024 + 255) / 256, 256, 0, stream>>>(x, xb, 8192 * 1024);
    csa_cvt_transpose<<<(1024 * 3072 + 255) / 256, 256, 0, stream>>>(w_attn, wqkvT, 1024, 3072);
    csa_cvt_transpose<<<(1024 * 1024 + 255) / 256, 256, 0, stream>>>(w_proj, wprojT, 1024, 1024);

    // qkv = x @ w_attn + b_attn, scattered into head-major Q/K/V^T
    csa_gemm_qkv<<<dim3(3072 / 64, 8192 / 128), 256, 0, stream>>>(xb, wqkvT, b_attn, Qb, Kb, Vt);

    // causal attention
    csa_attn<<<dim3(B_DIM * H_DIM, T_DIM / 128), 256, 0, stream>>>(Qb, Kb, Vt, attb);

    // out = att @ w_proj + b_proj
    csa_gemm_proj<<<dim3(1024 / 64, 8192 / 128), 256, 0, stream>>>(attb, wprojT, b_proj, out);
}
